// SigmoidAttention_31765578121786
// MI455X (gfx1250) — matrix operand order repaired and checked by reading, not compiled, not run on hardware
//
#include <hip/hip_runtime.h>
#include <hip/hip_bf16.h>

#define B_  2
#define S_  2048
#define D_  512
#define H_  8
#define HD_ 64

typedef __attribute__((ext_vector_type(16))) _Float16 v16h;
typedef __attribute__((ext_vector_type(8)))  _Float16 v8h;
typedef __attribute__((ext_vector_type(8)))  float    v8f;

// ---------- WMMA fragment helpers (contiguous k order, same for A and B, wave32) ----------

// A-matrix 16x32 f16: lane L<16 -> row L, K=0..15; lane L>=16 -> row L-16, K=16..31.
// `tile` points at (row0, k0) of the tile; ld = row stride in halfs.
__device__ __forceinline__ v16h load_a_frag(const _Float16* tile, int ld, int lane) {
    const int row = lane & 15;
    const int hi  = (lane & 16) ? 16 : 0;
    const _Float16* rp = tile + (size_t)row * ld + hi;
    return *(const v16h*)(rp);
}

// B-matrix 32x16 f16: lane L<16 -> col L, K=0..15; lane L>=16 -> col L-16, K=16..31.
// Source must be K-contiguous per column; ld = col stride in halfs.
__device__ __forceinline__ v16h load_b_frag(const _Float16* tile, int ld, int lane) {
    const int col = lane & 15;
    const int hi  = (lane & 16) ? 16 : 0;
    return *(const v16h*)(tile + (size_t)col * ld + hi);
}

__device__ __forceinline__ v8f wmma_f16(v16h a, v16h b, v8f c) {
    return __builtin_amdgcn_wmma_f32_16x16x32_f16(false, a, false, b, (short)0, c,
                                                  false, false);
}

// ---------- Stage 1: f32 -> f16 convert ----------
__global__ void sigattn_cvt_f16(const float* __restrict__ in,
                                _Float16* __restrict__ out, int n) {
    int i = blockIdx.x * blockDim.x + threadIdx.x;
    if (i < n) out[i] = (_Float16)in[i];
}

// ---------- Stage 2: QKV projection GEMM (y = x @ W^T + b) ----------
// Grid: (B*S/16, D/16, 3); block: 32 (one wave per 16x16 output tile).
// Q -> [B*H][S][HD] (scaled by 1/sqrt(HD)), K -> [B*H][S][HD], V -> [B*H][HD][S].
__global__ void __launch_bounds__(32)
sigattn_proj(const _Float16* __restrict__ xf,   // [B*S][D]
             const _Float16* __restrict__ wf,   // [3][D][D] (row = out feature)
             const float* __restrict__ bq,
             const float* __restrict__ bk,
             const float* __restrict__ bv,
             _Float16* __restrict__ qw,
             _Float16* __restrict__ kw,
             _Float16* __restrict__ vw) {
    const int lane = threadIdx.x;
    const int r0   = blockIdx.x * 16;            // token tile base
    const int o0   = blockIdx.y * 16;            // out-feature tile base
    const int mat  = blockIdx.z;                 // 0=Q 1=K 2=V
    const _Float16* wm = wf + (size_t)mat * D_ * D_;

    v8f acc = {};
#pragma unroll 4
    for (int kk = 0; kk < D_; kk += 32) {
        v16h a = load_a_frag(xf + (size_t)r0 * D_ + kk, D_, lane);  // x rows
        v16h b = load_b_frag(wm + (size_t)o0 * D_ + kk, D_, lane);  // W rows = B cols
        acc = wmma_f16(a, b, acc);
    }

    const int n = lane & 15;
    const int o = o0 + n;
    const float bias  = (mat == 0) ? bq[o] : (mat == 1) ? bk[o] : bv[o];
    const float scale = (mat == 0) ? 0.125f : 1.0f;   // fold 1/sqrt(64) into Q
    const int h = o / HD_, d = o % HD_;
    const int mofs = (lane & 16) ? 8 : 0;
#pragma unroll
    for (int r = 0; r < 8; ++r) {
        const int m  = r + mofs;
        const int t  = r0 + m;                 // global token
        const int bi = t / S_, si = t % S_;
        const int bh = bi * H_ + h;
        const _Float16 val = (_Float16)((acc[r] + bias) * scale);
        if (mat == 0)      qw[((size_t)bh * S_ + si) * HD_ + d] = val;
        else if (mat == 1) kw[((size_t)bh * S_ + si) * HD_ + d] = val;
        else               vw[((size_t)bh * HD_ + d) * S_ + si] = val;  // V transposed
    }
}

// ---------- Stage 3: fused sigmoid attention ----------
// Grid: B*H*(S/128) blocks of 128 threads (4 waves).
// Each wave owns a 32-row q-tile (two 16-row WMMA subtiles) so every K/V
// B-fragment load is amortized over 2 WMMAs: 16 WMMAs per 32-k iteration
// against 8 B-fragment loads.
__global__ void __launch_bounds__(128)
sigattn_attn(const _Float16* __restrict__ qw,   // [B*H][S][HD], pre-scaled
             const _Float16* __restrict__ kw,   // [B*H][S][HD]
             const _Float16* __restrict__ vw,   // [B*H][HD][S]
             float* __restrict__ out) {         // [B][S][D]
    __shared__ _Float16 lds[4][32 * 32];        // per-wave attn staging tile (2 KB)

    const int lane = threadIdx.x & 31;
    const int wave = threadIdx.x >> 5;
    const int qt   = blockIdx.x % (S_ / 128);
    const int bh   = blockIdx.x / (S_ / 128);
    const int q0   = qt * 128 + wave * 32;

    const _Float16* qp = qw + (size_t)bh * S_ * HD_;
    const _Float16* kp = kw + (size_t)bh * S_ * HD_;
    const _Float16* vp = vw + (size_t)bh * HD_ * S_;

    // Q tile A-fragments: 2 row-subtiles x 2 K-steps (HD=64), reused all k-tiles.
    v16h aq[2][2];
#pragma unroll
    for (int sub = 0; sub < 2; ++sub) {
        aq[sub][0] = load_a_frag(qp + (size_t)(q0 + sub * 16) * HD_ + 0,  HD_, lane);
        aq[sub][1] = load_a_frag(qp + (size_t)(q0 + sub * 16) * HD_ + 32, HD_, lane);
    }

    v8f acc[2][4] = {};
    _Float16* myl = lds[wave];
    const int n    = lane & 15;
    const int mofs = (lane & 16) ? 8 : 0;

    for (int k0 = 0; k0 < S_; k0 += 32) {
        // K B-fragments for columns [k0, k0+32): 2 col-blocks x 2 K-steps.
        v16h bk[2][2];
#pragma unroll
        for (int cb = 0; cb < 2; ++cb) {
            bk[cb][0] = load_b_frag(kp + (size_t)(k0 + cb * 16) * HD_ + 0,  HD_, lane);
            bk[cb][1] = load_b_frag(kp + (size_t)(k0 + cb * 16) * HD_ + 32, HD_, lane);
        }
        // ---- scores + sigmoid, staged to LDS as f16 [32 rows][32 cols] ----
#pragma unroll
        for (int sub = 0; sub < 2; ++sub) {
#pragma unroll
            for (int cb = 0; cb < 2; ++cb) {
                v8f s = {};
                s = wmma_f16(aq[sub][0], bk[cb][0], s);
                s = wmma_f16(aq[sub][1], bk[cb][1], s);
#pragma unroll
                for (int r = 0; r < 8; ++r) {
                    const float x = s[r];
                    const float e = __builtin_amdgcn_exp2f(x * -1.44269504088896f);
                    const float a = 1.0f / (1.0f + e);
                    myl[(sub * 16 + r + mofs) * 32 + cb * 16 + n] = (_Float16)a;
                }
            }
        }
        // Wave-private LDS tile: DS ops are in-order per wave, just drain DScnt.
        __builtin_amdgcn_wave_barrier();
        asm volatile("s_wait_dscnt 0" ::: "memory");
        __builtin_amdgcn_wave_barrier();

        v16h aa[2];
        aa[0] = load_a_frag(myl,           32, lane);   // attn rows 0..15
        aa[1] = load_a_frag(myl + 16 * 32, 32, lane);   // attn rows 16..31

        // ---- out += attn @ V over this 32-k chunk, 4 HD/16 column tiles ----
#pragma unroll
        for (int dt = 0; dt < 4; ++dt) {
            v16h bvf = load_b_frag(vp + (size_t)(dt * 16) * S_ + k0, S_, lane);
            acc[0][dt] = wmma_f16(aa[0], bvf, acc[0][dt]);
            acc[1][dt] = wmma_f16(aa[1], bvf, acc[1][dt]);
        }
    }

    // ---- merge heads and store f32 output [B][S][D] ----
    const int b = bh / H_, h = bh % H_;
#pragma unroll
    for (int sub = 0; sub < 2; ++sub) {
#pragma unroll
        for (int dt = 0; dt < 4; ++dt) {
#pragma unroll
            for (int r = 0; r < 8; ++r) {
                const int m = sub * 16 + r + mofs;
                out[((size_t)b * S_ + (q0 + m)) * D_ + h * HD_ + dt * 16 + n] =
                    acc[sub][dt][r];
            }
        }
    }
}

// ---------- Host launch ----------
extern "C" void kernel_launch(void* const* d_in, const int* in_sizes, int n_in,
                              void* d_out, int out_size, void* d_ws, size_t ws_size,
                              hipStream_t stream) {
    const float* x  = (const float*)d_in[0];
    const float* wq = (const float*)d_in[1];
    const float* bq = (const float*)d_in[2];
    const float* wk = (const float*)d_in[3];
    const float* bk = (const float*)d_in[4];
    const float* wv = (const float*)d_in[5];
    const float* bv = (const float*)d_in[6];
    float* out = (float*)d_out;

    char* ws = (char*)d_ws;
    size_t off = 0;
    auto alloc = [&](size_t bytes) -> char* {
        char* p = ws + off;
        off = (off + bytes + 255) & ~(size_t)255;
        return p;
    };
    _Float16* xf = (_Float16*)alloc((size_t)B_ * S_ * D_ * 2);
    _Float16* wf = (_Float16*)alloc((size_t)3 * D_ * D_ * 2);
    _Float16* qw = (_Float16*)alloc((size_t)B_ * H_ * S_ * HD_ * 2);
    _Float16* kw = (_Float16*)alloc((size_t)B_ * H_ * S_ * HD_ * 2);
    _Float16* vw = (_Float16*)alloc((size_t)B_ * H_ * S_ * HD_ * 2);
    (void)ws_size; (void)in_sizes; (void)n_in; (void)out_size;

    // Stage 1: converts
    const int nx = B_ * S_ * D_;
    const int nw = D_ * D_;
    sigattn_cvt_f16<<<(nx + 255) / 256, 256, 0, stream>>>(x,  xf,          nx);
    sigattn_cvt_f16<<<(nw + 255) / 256, 256, 0, stream>>>(wq, wf,          nw);
    sigattn_cvt_f16<<<(nw + 255) / 256, 256, 0, stream>>>(wk, wf + nw,     nw);
    sigattn_cvt_f16<<<(nw + 255) / 256, 256, 0, stream>>>(wv, wf + 2 * nw, nw);

    // Stage 2: QKV projection
    dim3 pg(B_ * S_ / 16, D_ / 16, 3);
    sigattn_proj<<<pg, 32, 0, stream>>>(xf, wf, bq, bk, bv, qw, kw, vw);

    // Stage 3: fused sigmoid attention
    sigattn_attn<<<B_ * H_ * (S_ / 128), 128, 0, stream>>>(qw, kw, vw, out);
}
